// VIVQ_48163763257766
// MI455X (gfx1250) — compile-verified, matrix operand order checked by reading, not run on hardware
//
#include <hip/hip_runtime.h>
#include <hip/hip_bf16.h>

typedef __bf16 bf16;
typedef __attribute__((ext_vector_type(16))) __bf16 v16bf;
typedef __attribute__((ext_vector_type(8)))  __bf16 v8bf;
typedef __attribute__((ext_vector_type(8)))  float  v8f;

__device__ __forceinline__ float gelu_f(float x) {
  return 0.5f * x * (1.0f + erff(x * 0.70710678118654752440f));
}
__device__ __forceinline__ int iclamp(int v, int lo, int hi) {
  return v < lo ? lo : (v > hi ? hi : v);
}

// ---------------------------------------------------------------------------
// WMMA GEMM: out[M,N] = A[M,Kp] * Wb[N,Kp]^T + bias (+ res)
// A is pre-activated bf16 (GELU fused into the producer pass), Kp-padded.
// Block tile 128x128, 8 waves (wave32) as 4(M)x2(N); each wave computes a
// 32x64 patch = 2 A-frags x 4 B-frags = 8x v_wmma_f32_16x16x32_bf16 per
// K-step of 32. All staging is b128 vector traffic (global->LDS->frags).
// ---------------------------------------------------------------------------
#define GBM 128
#define GBN 128
#define GBK 32
#define GLDA 40   // bf16 elems per LDS row (32 + pad); 80B rows keep 16B chunks
#define GLDB 40

__device__ __forceinline__ v16bf frag_ld(const bf16* base, int lane, int ld) {
  // 16-bit operand layout (ISA 7.12.2): lanes 0-15 -> row, K {0..7,16..23};
  // lanes 16-31 -> same rows, K {8..15,24..31}.
  const int row = lane & 15;
  const int hi  = (lane >> 4) & 1;
  const bf16* p = base + row * ld + hi * 8;
  v8bf lo  = *(const v8bf*)(p);
  v8bf hi8 = *(const v8bf*)(p + 16);
  v16bf r;
#pragma unroll
  for (int i = 0; i < 8; ++i) { r[i] = lo[i]; r[i + 8] = hi8[i]; }
  return r;
}

__global__ __launch_bounds__(256) void gemm_bf16_wmma(
    const bf16* __restrict__ A,   // [M, Kp] pre-activated
    const bf16* __restrict__ Wb,  // [N, Kp] packed weights
    int Kp,
    const float* __restrict__ bias,
    const float* __restrict__ res,
    float* __restrict__ out,
    int M, int N)
{
  __shared__ __align__(16) bf16 sA[GBM * GLDA];
  __shared__ __align__(16) bf16 sB[GBN * GLDB];

  const int tid  = threadIdx.x;
  const int lane = tid & 31;
  const int wave = tid >> 5;
  const int wm = wave & 3;   // 4 wave-rows * 32 = 128 M
  const int wn = wave >> 2;  // 2 wave-cols * 64 = 128 N
  const int m_blk = blockIdx.x * GBM;
  const int n_blk = blockIdx.y * GBN;

  v8f acc[8] = {};

  const int srow  = tid >> 1;        // 0..127
  const int shalf = (tid & 1) * 16;  // 0 or 16

  v8bf zv;
#pragma unroll
  for (int i = 0; i < 8; ++i) zv[i] = (bf16)0.f;

  for (int kt = 0; kt < Kp; kt += GBK) {
    { // stage A tile: two b128 loads -> two b128 LDS stores
      const int m = m_blk + srow;
      v8bf c0 = zv, c1 = zv;
      if (m < M) {
        const bf16* p = A + (size_t)m * Kp + kt + shalf;
        c0 = *(const v8bf*)p;
        c1 = *(const v8bf*)(p + 8);
      }
      *(v8bf*)&sA[srow * GLDA + shalf]     = c0;
      *(v8bf*)&sA[srow * GLDA + shalf + 8] = c1;
    }
    { // stage B tile
      const int n = n_blk + srow;
      v8bf c0 = zv, c1 = zv;
      if (n < N) {
        const bf16* p = Wb + (size_t)n * Kp + kt + shalf;
        c0 = *(const v8bf*)p;
        c1 = *(const v8bf*)(p + 8);
      }
      *(v8bf*)&sB[srow * GLDB + shalf]     = c0;
      *(v8bf*)&sB[srow * GLDB + shalf + 8] = c1;
    }
    __syncthreads();

    v16bf a0 = frag_ld(&sA[(wm * 32     ) * GLDA], lane, GLDA);
    v16bf a1 = frag_ld(&sA[(wm * 32 + 16) * GLDA], lane, GLDA);
    v16bf b0 = frag_ld(&sB[(wn * 64     ) * GLDB], lane, GLDB);
    v16bf b1 = frag_ld(&sB[(wn * 64 + 16) * GLDB], lane, GLDB);
    v16bf b2 = frag_ld(&sB[(wn * 64 + 32) * GLDB], lane, GLDB);
    v16bf b3 = frag_ld(&sB[(wn * 64 + 48) * GLDB], lane, GLDB);

    acc[0] = __builtin_amdgcn_wmma_f32_16x16x32_bf16(false, a0, false, b0, (short)0, acc[0], false, false);
    acc[1] = __builtin_amdgcn_wmma_f32_16x16x32_bf16(false, a0, false, b1, (short)0, acc[1], false, false);
    acc[2] = __builtin_amdgcn_wmma_f32_16x16x32_bf16(false, a0, false, b2, (short)0, acc[2], false, false);
    acc[3] = __builtin_amdgcn_wmma_f32_16x16x32_bf16(false, a0, false, b3, (short)0, acc[3], false, false);
    acc[4] = __builtin_amdgcn_wmma_f32_16x16x32_bf16(false, a1, false, b0, (short)0, acc[4], false, false);
    acc[5] = __builtin_amdgcn_wmma_f32_16x16x32_bf16(false, a1, false, b1, (short)0, acc[5], false, false);
    acc[6] = __builtin_amdgcn_wmma_f32_16x16x32_bf16(false, a1, false, b2, (short)0, acc[6], false, false);
    acc[7] = __builtin_amdgcn_wmma_f32_16x16x32_bf16(false, a1, false, b3, (short)0, acc[7], false, false);

    __syncthreads();
  }

  // C/D layout: VGPR r, lanes 0-15 -> M=r, N=lane; lanes 16-31 -> M=r+8, N=lane-16
  const int col = lane & 15;
  const int sel = lane >> 4;
#pragma unroll
  for (int ij = 0; ij < 8; ++ij) {
    const int i = ij >> 2, j = ij & 3;
    const int nb = n_blk + wn * 64 + j * 16 + col;
#pragma unroll
    for (int r = 0; r < 8; ++r) {
      const int mb = m_blk + wm * 32 + i * 16 + r + sel * 8;
      if (mb < M && nb < N) {
        float v = acc[ij][r] + bias[nb];
        if (res) v += res[(size_t)mb * N + nb];
        out[(size_t)mb * N + nb] = v;
      }
    }
  }
}

// ------------------------- support kernels ---------------------------------

// Pre-activation pass: dst[M,Kp](bf16) = (gelu?)(concat(s1[M,K1], s2[M,K2])),
// zero-padded to Kp. Each element GELU'd exactly once (vs once per N-tile).
__global__ void act_concat_bf16(const float* __restrict__ s1, int K1,
                                const float* __restrict__ s2, int K2,
                                bf16* __restrict__ dst, int M, int Kp, int do_gelu)
{
  size_t idx = (size_t)blockIdx.x * blockDim.x + threadIdx.x;
  size_t total = (size_t)M * Kp;
  if (idx >= total) return;
  int k = (int)(idx % Kp);
  size_t m = idx / Kp;
  float v = 0.f;
  if (k < K1)           v = s1[m * K1 + k];
  else if (k < K1 + K2) v = s2[m * K2 + (k - K1)];
  if (do_gelu) v = gelu_f(v);
  dst[idx] = (bf16)v;
}

__global__ void pack_w_bf16(const float* __restrict__ w, bf16* __restrict__ dst,
                            int N, int K, int Kp) {
  int idx = blockIdx.x * blockDim.x + threadIdx.x;
  if (idx >= N * Kp) return;
  int n = idx / Kp, k = idx - n * Kp;
  dst[idx] = (bf16)((k < K) ? w[(size_t)n * K + k] : 0.f);
}

// depthwise 3x3x3, edge ("clamp") padding, GELU applied on input, bias added.
__global__ void dwconv3_gelu(const float* __restrict__ in, const float* __restrict__ wd,
                             const float* __restrict__ bd, float* __restrict__ out,
                             int T, int H, int W, int C)
{
  size_t idx = (size_t)blockIdx.x * blockDim.x + threadIdx.x;
  size_t total = (size_t)2 * T * H * W * C;
  if (idx >= total) return;
  int c = (int)(idx % C);
  size_t sp = idx / C;
  int x = (int)(sp % W); sp /= W;
  int y = (int)(sp % H); sp /= H;
  int t = (int)(sp % T);
  int b = (int)(sp / T);
  float acc = bd[c];
  const float* wc = wd + (size_t)c * 27;
  int wi = 0;
  for (int dt = -1; dt <= 1; ++dt) {
    int tt = iclamp(t + dt, 0, T - 1);
    for (int dy = -1; dy <= 1; ++dy) {
      int yy = iclamp(y + dy, 0, H - 1);
      for (int dx = -1; dx <= 1; ++dx) {
        int xx = iclamp(x + dx, 0, W - 1);
        float v = in[((((size_t)b * T + tt) * H + yy) * W + xx) * C + c];
        acc += gelu_f(v) * wc[wi++];
      }
    }
  }
  out[idx] = acc;
}

// im2col for the stem conv (stride 2,4,4 / kernel 2,4,4), building the
// frame|image|video time-concat on the fly. A[m, k]: m=((b*6+t)*32+y)*32+x,
// k = ci*32 + dt*16 + dy*4 + dx  (matches OIDHW flatten of stem_w).
__global__ void stem_im2col(const float* __restrict__ image, const float* __restrict__ video,
                            const float* __restrict__ frame, float* __restrict__ A)
{
  int idx = blockIdx.x * blockDim.x + threadIdx.x;
  const int M = 12288, Kc = 96;
  if (idx >= M * Kc) return;
  int m = idx / Kc, k = idx - m * Kc;
  int ci = k >> 5; int r = k & 31;
  int dt = r >> 4; int dy = (r >> 2) & 3; int dx = r & 3;
  int x = m & 31; int y = (m >> 5) & 31; int t = (m >> 10) % 6; int b = m / 6144;
  int tt = 2 * t + dt, sy = 4 * y + dy, sx = 4 * x + dx;
  float v;
  if (tt == 0)      v = frame[((size_t)ci * 128 + sy) * 128 + sx];
  else if (tt == 1) v = image[(((size_t)b * 3 + ci) * 128 + sy) * 128 + sx];
  else              v = video[((((size_t)b * 10 + (tt - 2)) * 3 + ci) * 128 + sy) * 128 + sx];
  A[idx] = v;
}

// spatial ::2 downsample of [2,6,32,32,128] -> [2,6,16,16,128]
__global__ void down2_k(const float* __restrict__ in, float* __restrict__ out) {
  int idx = blockIdx.x * blockDim.x + threadIdx.x;
  if (idx >= 2 * 6 * 16 * 16 * 128) return;
  int c = idx & 127; int r = idx >> 7;
  int x = r & 15; r >>= 4; int y = r & 15; r >>= 4;
  int t = r % 6; int b = r / 6;
  out[idx] = in[((((size_t)b * 6 + t) * 32 + 2 * y) * 32 + 2 * x) * 128 + c];
}

// pixel-unshuffle: [2,6,32,32,128] -> [2,6,16,16,512], oc = c*4 + i*2 + j
__global__ void pxunshuf_k(const float* __restrict__ in, float* __restrict__ out) {
  int idx = blockIdx.x * blockDim.x + threadIdx.x;
  if (idx >= 2 * 6 * 16 * 16 * 512) return;
  int oc = idx & 511; int r = idx >> 9;
  int x = r & 15; r >>= 4; int y = r & 15; r >>= 4;
  int t = r % 6; int b = r / 6;
  int c = oc >> 2; int i = (oc >> 1) & 1; int j = oc & 1;
  out[idx] = in[((((size_t)b * 6 + t) * 32 + 2 * y + i) * 32 + (2 * x + j)) * 128 + c];
}

// nearest-neighbor upsample with optional leading-time slice via toff:
// out[b,t,y,x,c] = in[b,(t+toff)/ft, y/fh, x/fw, c]
__global__ void upsample_k(const float* __restrict__ in, float* __restrict__ out,
                           int Tin, int Hin, int Win, int Tout, int Hout, int Wout,
                           int C, int ft, int fh, int fw, int toff)
{
  size_t idx = (size_t)blockIdx.x * blockDim.x + threadIdx.x;
  size_t total = (size_t)2 * Tout * Hout * Wout * C;
  if (idx >= total) return;
  int c = (int)(idx % C); size_t r = idx / C;
  int x = (int)(r % Wout); r /= Wout;
  int y = (int)(r % Hout); r /= Hout;
  int t = (int)(r % Tout); int b = (int)(r / Tout);
  int ti = (t + toff) / ft, yi = y / fh, xi = x / fw;
  out[idx] = in[((((size_t)b * Tin + ti) * Hin + yi) * Win + xi) * C + c];
}

__global__ void bn_stats_k(const float* __restrict__ z, float* __restrict__ stats, int M) {
  __shared__ float ss[512], sq[512];
  int tid = threadIdx.x;
  int c = tid & 15;
  float s = 0.f, q = 0.f;
  for (int m = tid >> 4; m < M; m += 32) { float v = z[(size_t)m * 16 + c]; s += v; q += v * v; }
  ss[tid] = s; sq[tid] = q; __syncthreads();
  for (int off = 256; off >= 16; off >>= 1) {
    if (tid < off) { ss[tid] += ss[tid + off]; sq[tid] += sq[tid + off]; }
    __syncthreads();
  }
  if (tid < 16) {
    float mean = ss[tid] / (float)M;
    float var  = sq[tid] / (float)M - mean * mean;
    stats[tid] = mean; stats[16 + tid] = var;
  }
}

__global__ void bn_apply_k(float* __restrict__ z, const float* __restrict__ stats,
                           const float* __restrict__ g, const float* __restrict__ b, int M) {
  int idx = blockIdx.x * blockDim.x + threadIdx.x;
  if (idx >= M * 16) return;
  int c = idx & 15;
  float v = z[idx];
  z[idx] = (v - stats[c]) * rsqrtf(stats[16 + c] + 1e-5f) * g[c] + b[c];
}

// VQ: nearest codebook row (L2, first-min tie-break like argmin), commit sum.
__global__ __launch_bounds__(256) void vq_k(const float* __restrict__ z,
                                            const float* __restrict__ cb,
                                            float* __restrict__ q,
                                            float* __restrict__ commit_acc, int M)
{
  __shared__ float scb[256 * 16];
  __shared__ float red[256];
  int tid = threadIdx.x;
  int m = blockIdx.x * 256 + tid;
  float x[16];
#pragma unroll
  for (int i = 0; i < 16; ++i) x[i] = (m < M) ? z[(size_t)m * 16 + i] : 0.f;
  float best = 3.4e38f; int bi = 0;
  for (int ch = 0; ch < 4; ++ch) {
#pragma unroll
    for (int i = 0; i < 16; ++i) scb[tid * 16 + i] = cb[((size_t)ch * 256 + tid) * 16 + i];
    __syncthreads();
    for (int j = 0; j < 256; ++j) {
      float d = 0.f;
#pragma unroll
      for (int i = 0; i < 16; ++i) { float df = x[i] - scb[j * 16 + i]; d += df * df; }
      if (d < best) { best = d; bi = ch * 256 + j; }
    }
    __syncthreads();
  }
  float cs = 0.f;
  if (m < M) {
#pragma unroll
    for (int i = 0; i < 16; ++i) {
      float qv = cb[(size_t)bi * 16 + i];
      q[(size_t)m * 16 + i] = qv;
      float df = qv - x[i];
      cs += df * df;
    }
  }
  red[tid] = cs; __syncthreads();
  for (int off = 128; off > 0; off >>= 1) {
    if (tid < off) red[tid] += red[tid + off];
    __syncthreads();
  }
  if (tid == 0) atomicAdd(commit_acc, red[0]);
}

__global__ void zero_k(float* p) { p[0] = 0.f; }
__global__ void write_commit_k(const float* acc, float* out) { out[0] = acc[0] * (1.f / 49152.f); }

// out = sigmoid(up4(o0) + up2(o1) + o2 + o3), layout (B,11,3,128,128)
__global__ void final_out_k(const float* __restrict__ o0, const float* __restrict__ o1,
                            const float* __restrict__ o2, const float* __restrict__ o3,
                            float* __restrict__ out)
{
  int idx = blockIdx.x * blockDim.x + threadIdx.x;
  if (idx >= 2 * 11 * 128 * 128) return;
  int x = idx & 127; int r = idx >> 7;
  int y = r & 127; r >>= 7;
  int t = r % 11; int b = r / 11;
  size_t bt = (size_t)b * 11 + t;
#pragma unroll
  for (int cc = 0; cc < 3; ++cc) {
    float v = o0[((bt * 32 + (y >> 2)) * 32 + (x >> 2)) * 3 + cc]
            + o1[((bt * 64 + (y >> 1)) * 64 + (x >> 1)) * 3 + cc]
            + o2[((bt * 128 + y) * 128 + x) * 3 + cc]
            + o3[((bt * 128 + y) * 128 + x) * 3 + cc];
    out[((bt * 3 + cc) * 128 + (size_t)y) * 128 + x] = 1.f / (1.f + expf(-v));
  }
}

// --------------------------- host orchestration ----------------------------

struct RBP { const float *w1, *b1, *wd, *bd, *w2, *b2; };

extern "C" void kernel_launch(void* const* d_in, const int* in_sizes, int n_in,
                              void* d_out, int out_size, void* d_ws, size_t ws_size,
                              hipStream_t stream)
{
  (void)in_sizes; (void)n_in; (void)ws_size;
  int ci = 0;
  auto NX = [&]() { return (const float*)d_in[ci++]; };
  const float* image = NX();
  const float* video = NX();
  const float* frame = NX();
  const float* stem_w = NX(); const float* stem_b = NX();
  auto RBGET = [&]() { RBP r; r.w1 = NX(); r.b1 = NX(); r.wd = NX(); r.bd = NX(); r.w2 = NX(); r.b2 = NX(); return r; };
  RBP enc0 = RBGET(), enc1 = RBGET();
  RBP encb[8]; for (int i = 0; i < 8; ++i) encb[i] = RBGET();
  const float* map_w = NX(); const float* map_b = NX();
  const float* bn_g = NX();  const float* bn_b = NX();
  const float* cb = NX();
  const float* un_w = NX();  const float* un_b = NX();
  RBP decb[8]; for (int i = 0; i < 8; ++i) decb[i] = RBGET();
  RBP dec[8];  for (int i = 0; i < 8; ++i) dec[i]  = RBGET();
  const float* sc1w = NX(); const float* sc1b = NX();
  const float* sc3w = NX(); const float* sc3b = NX();
  const float* sc5w = NX(); const float* sc5b = NX();
  const float* ocw[4]; const float* ocb[4];
  for (int i = 0; i < 4; ++i) { ocw[i] = NX(); ocb[i] = NX(); }

  char* base = (char*)d_ws; size_t off = 0;
  auto ALLOCB = [&](size_t bytes) -> void* {
    void* p = (void*)(base + off);
    off += ((bytes + 255) & ~(size_t)255);
    return p;
  };
  auto ALLOC = [&](size_t elems) -> float* { return (float*)ALLOCB(elems * sizeof(float)); };

  const int M16 = 3072, Mst = 12288, M32 = 22528, M64 = 90112, M128 = 360448;
  float* Astem = ALLOC((size_t)Mst * 96);
  float* X0    = ALLOC((size_t)Mst * 128);
  float* S0    = ALLOC((size_t)M16 * 128);
  float* X512  = ALLOC((size_t)M16 * 512);   // main 512-ch buffer, enc then dec
  float* Z     = ALLOC((size_t)M16 * 16);
  float* Q     = ALLOC((size_t)M16 * 16);
  float* stats = ALLOC(32);
  float* commit = ALLOC(8);
  float* Ha = ALLOC(11534336);               // hidden ping (max M*Hh)
  float* Hb = ALLOC(11534336);               // hidden pong
  float* Sa = ALLOC(2883584);                // small skip ping
  float* Sb = ALLOC(11534336);               // big (upsampled) skip
  float* Y128 = ALLOC((size_t)M32 * 128);
  float* Y32  = ALLOC((size_t)M64 * 32);
  float* Y8   = ALLOC((size_t)M128 * 8);
  float* Tsc  = ALLOC(720896);               // shortcut-conv temp
  float* O0 = ALLOC((size_t)M32 * 3);
  float* O1 = ALLOC((size_t)M64 * 3);
  float* O2 = ALLOC((size_t)M128 * 3);
  float* O3 = ALLOC((size_t)M128 * 3);
  bf16* wpack = (bf16*)ALLOCB((size_t)2048 * 640 * sizeof(bf16));
  bf16* Abf   = (bf16*)ALLOCB((size_t)23068672 * sizeof(bf16)); // max M*Kp (360448*64)

  auto GEMM = [&](const float* A1, int K1, const float* A2, int K2,
                  const float* w, const float* b, const float* res, float* out,
                  int M, int N, int geluin) {
    int K = K1 + K2;
    int Kp = (K + 31) & ~31;
    size_t atot = (size_t)M * Kp;
    act_concat_bf16<<<(unsigned)((atot + 255) / 256), 256, 0, stream>>>(
        A1, K1, A2, K2, Abf, M, Kp, geluin);
    int wtot = N * Kp;
    pack_w_bf16<<<(wtot + 255) / 256, 256, 0, stream>>>(w, wpack, N, K, Kp);
    dim3 g((M + GBM - 1) / GBM, (N + GBN - 1) / GBN);
    gemm_bf16_wmma<<<g, 256, 0, stream>>>(Abf, wpack, Kp, b, res, out, M, N);
  };
  auto DW = [&](const float* h1, const float* wd, const float* bd, float* h2,
                int T, int H, int W, int C) {
    size_t tot = (size_t)2 * T * H * W * C;
    dwconv3_gelu<<<(unsigned)((tot + 255) / 256), 256, 0, stream>>>(h1, wd, bd, h2, T, H, W, C);
  };
  auto RBRUN = [&](float* x, int C, const float* s, int Cs, const RBP& p, int Hh,
                   int T, int H, int W) {
    int M = 2 * T * H * W;
    GEMM(x, C, s, Cs, p.w1, p.b1, nullptr, Ha, M, Hh, 1);   // gelu(concat) -> 1x1
    DW(Ha, p.wd, p.bd, Hb, T, H, W, Hh);                    // gelu -> edge-pad dw3x3x3
    GEMM(Hb, Hh, nullptr, 0, p.w2, p.b2, x, x, M, C, 1);    // gelu -> 1x1 + residual
  };
  auto UP = [&](const float* in, float* out, int Tin, int Hin, int Win,
                int Tout, int Hout, int Wout, int C, int ft, int fh, int fw, int toff) {
    size_t tot = (size_t)2 * Tout * Hout * Wout * C;
    upsample_k<<<(unsigned)((tot + 255) / 256), 256, 0, stream>>>(
        in, out, Tin, Hin, Win, Tout, Hout, Wout, C, ft, fh, fw, toff);
  };
  auto COPY = [&](float* dst, const float* src, size_t elems) {
    hipMemcpyAsync(dst, src, elems * sizeof(float), hipMemcpyDeviceToDevice, stream);
  };

  // ----------------- encoder -----------------
  stem_im2col<<<(Mst * 96 + 255) / 256, 256, 0, stream>>>(image, video, frame, Astem);
  GEMM(Astem, 96, nullptr, 0, stem_w, stem_b, nullptr, X0, Mst, 128, 0);
  down2_k<<<(M16 * 128 + 255) / 256, 256, 0, stream>>>(X0, S0);           // prev_s
  RBRUN(X0, 128, nullptr, 0, enc0, 512, 6, 32, 32);
  pxunshuf_k<<<(M16 * 512 + 255) / 256, 256, 0, stream>>>(X0, X512);
  RBRUN(X512, 512, S0, 128, enc1, 2048, 6, 16, 16);
  for (int i = 0; i < 8; ++i) RBRUN(X512, 512, nullptr, 0, encb[i], 2048, 6, 16, 16);

  // ----------------- quantize -----------------
  GEMM(X512, 512, nullptr, 0, map_w, map_b, nullptr, Z, M16, 16, 0);
  bn_stats_k<<<1, 512, 0, stream>>>(Z, stats, M16);
  bn_apply_k<<<(M16 * 16 + 255) / 256, 256, 0, stream>>>(Z, stats, bn_g, bn_b, M16);
  zero_k<<<1, 1, 0, stream>>>(commit);
  vq_k<<<(M16 + 255) / 256, 256, 0, stream>>>(Z, cb, Q, commit, M16);
  GEMM(Q, 16, nullptr, 0, un_w, un_b, nullptr, X512, M16, 512, 0);
  for (int i = 0; i < 8; ++i) RBRUN(X512, 512, nullptr, 0, decb[i], 2048, 6, 16, 16);

  // ----------------- decoder -----------------
  COPY(Sa, X512, (size_t)M16 * 512);                         // s for dec[1]
  RBRUN(X512, 512, nullptr, 0, dec[0], 2048, 6, 16, 16);
  UP(X512, Sb, 6, 16, 16, 11, 32, 32, 512, 2, 2, 2, 1);      // prev_s (t-sliced)
  RBRUN(X512, 512, Sa, 512, dec[1], 2048, 6, 16, 16);
  GEMM(X512, 512, nullptr, 0, sc1w, sc1b, nullptr, Tsc, M16, 128, 0);  // conv∘up = up∘conv
  UP(Tsc, Y128, 6, 16, 16, 11, 32, 32, 128, 2, 2, 2, 1);
  GEMM(Y128, 128, nullptr, 0, ocw[0], ocb[0], nullptr, O0, M32, 3, 0);

  COPY(Sa, Y128, (size_t)M32 * 128);
  RBRUN(Y128, 128, Sb, 512, dec[2], 512, 11, 32, 32);
  UP(Y128, Sb, 11, 32, 32, 11, 64, 64, 128, 1, 2, 2, 0);
  RBRUN(Y128, 128, Sa, 128, dec[3], 512, 11, 32, 32);
  GEMM(Y128, 128, nullptr, 0, sc3w, sc3b, nullptr, Tsc, M32, 32, 0);
  UP(Tsc, Y32, 11, 32, 32, 11, 64, 64, 32, 1, 2, 2, 0);
  GEMM(Y32, 32, nullptr, 0, ocw[1], ocb[1], nullptr, O1, M64, 3, 0);

  COPY(Sa, Y32, (size_t)M64 * 32);
  RBRUN(Y32, 32, Sb, 128, dec[4], 128, 11, 64, 64);
  UP(Y32, Sb, 11, 64, 64, 11, 128, 128, 32, 1, 2, 2, 0);
  RBRUN(Y32, 32, Sa, 32, dec[5], 128, 11, 64, 64);
  GEMM(Y32, 32, nullptr, 0, sc5w, sc5b, nullptr, Tsc, M64, 8, 0);
  UP(Tsc, Y8, 11, 64, 64, 11, 128, 128, 8, 1, 2, 2, 0);
  GEMM(Y8, 8, nullptr, 0, ocw[2], ocb[2], nullptr, O2, M128, 3, 0);

  COPY(Sa, Y8, (size_t)M128 * 8);
  RBRUN(Y8, 8, Sb, 32, dec[6], 32, 11, 128, 128);
  RBRUN(Y8, 8, Sa, 8, dec[7], 32, 11, 128, 128);
  GEMM(Y8, 8, nullptr, 0, ocw[3], ocb[3], nullptr, O3, M128, 3, 0);

  float* outF = (float*)d_out;
  final_out_k<<<(M128 + 255) / 256, 256, 0, stream>>>(O0, O1, O2, O3, outF);
  write_commit_k<<<1, 1, 0, stream>>>(commit, outF + (out_size - 1));
}
